// BiGRUEncoder_19988777796328
// MI455X (gfx1250) — matrix operand order repaired and checked by reading, not compiled, not run on hardware
//
#include <hip/hip_runtime.h>
#include <hip/hip_bf16.h>

typedef __attribute__((ext_vector_type(16))) __bf16 v16bf;
typedef __attribute__((ext_vector_type(8)))  __bf16 v8bf;
typedef __attribute__((ext_vector_type(8)))  float  v8f;
typedef __attribute__((ext_vector_type(4)))  unsigned int v4u;
typedef __attribute__((ext_vector_type(2)))  unsigned int v2u;

#define B_TOT 8192
#define T_WIN 15
#define D_IN  512
#define MAXL  8
#define MTILE 32

static __device__ __forceinline__ unsigned short f2bf(float f) {
  union { float f; unsigned u; } c; c.f = f;
  unsigned u = c.u;
  u += 0x7FFFu + ((u >> 16) & 1u);          // round-to-nearest-even
  return (unsigned short)(u >> 16);
}

static __device__ __forceinline__ v16bf cat16(v8bf a, v8bf b) {
  return __builtin_shufflevector(a, b, 0,1,2,3,4,5,6,7,8,9,10,11,12,13,14,15);
}

// A fragment (16x32 bf16, M x K) from row-major storage, row stride RS elems.
// lane: row = (Msub*16 + (lane&15)); kbase = k0 + (lane>>4)*8.
// Per ISA: lanes 0-15 hold K {0..7, 16..23}, lanes 16-31 hold K {8..15, 24..31}.
static __device__ __forceinline__ v16bf loadA(const unsigned short* p, int row, int RS, int kbase) {
  const unsigned short* q = p + row * RS + kbase;
  v8bf lo = *(const v8bf*)(q);
  v8bf hi = *(const v8bf*)(q + 16);
  return cat16(lo, hi);
}

// B fragment (32x16 bf16, K x N) from row-major weight W[N][K], row stride KS.
// lane holds column 'nrow'; kbase = k0 + (lane>>4)*8; K {kbase..+7, kbase+16..+23}.
static __device__ __forceinline__ v16bf loadB(const unsigned short* W, int nrow, int KS, int kbase) {
  const unsigned short* q = W + (size_t)nrow * KS + kbase;
  v8bf lo = *(const v8bf*)(q);
  v8bf hi = *(const v8bf*)(q + 16);
  return cat16(lo, hi);
}

static __device__ __forceinline__ v8f wmma_bf(v16bf a, v16bf b, v8f c) {
  return __builtin_amdgcn_wmma_f32_16x16x32_bf16(false, a, false, b, (short)0, c, false, false);
}

static __device__ __forceinline__ v8f splat8(float v) {
  v8f r;
  #pragma unroll
  for (int i = 0; i < 8; ++i) r[i] = v;
  return r;
}

static __device__ __forceinline__ float sigmf(float x) { return 1.0f / (1.0f + __expf(-x)); }

__global__ void cvt_bf16_kernel(const float* __restrict__ src, unsigned short* __restrict__ dst, int n) {
  int i = blockIdx.x * blockDim.x + threadIdx.x;
  if (i < n) dst[i] = f2bf(src[i]);
}

__global__ __launch_bounds__(256)
void bigru_fused(const float* __restrict__ pw, const int* __restrict__ wlen,
                 const unsigned short* __restrict__ Wihf, const unsigned short* __restrict__ Whhf,
                 const unsigned short* __restrict__ Wihb, const unsigned short* __restrict__ Whhb,
                 const unsigned short* __restrict__ W1b,  const unsigned short* __restrict__ W2b,
                 const float* __restrict__ bihf, const float* __restrict__ bhhf,
                 const float* __restrict__ bihb, const float* __restrict__ bhhb,
                 const float* __restrict__ b1,   const float* __restrict__ b2,
                 float* __restrict__ out) {
  __shared__ unsigned short sm[49152];   // 96KB: xs | hbf0 | hbf1 (GRU), comb | hid (MLP)
  __shared__ unsigned short hfb[16384];  // 32KB: bf16 forward final hidden
  __shared__ int lenF_s[32], lenB_s[32];

  unsigned short* xs   = sm;             // 32 x 512 bf16  (x_t tile)
  unsigned short* hbf0 = sm + 16384;     // 32 x 512 bf16  (h ping)
  unsigned short* hbf1 = sm + 32768;     // 32 x 512 bf16  (h pong)
  unsigned short* comb = sm;             // 32 x 1024 bf16 (aliases xs+hbf0, MLP stage)
  unsigned short* hid  = sm + 32768;     // 32 x 512 bf16  (aliases hbf1, MLP stage)

  const int tid   = threadIdx.x;
  const int wave  = tid >> 5;
  const int lane  = tid & 31;
  const int laneN = lane & 15;
  const int hi8   = (lane >> 4) * 8;
  const int hi16  = (lane >> 4) * 16;
  const int gb    = blockIdx.x * MTILE;

  if (tid < 32) {
    int wl = wlen[gb + tid];
    lenF_s[tid] = ((wl - 1) >> 1) + 1;
    lenB_s[tid] = (wl >> 1) + 1;
  }

  float hreg[4][2][8];   // persistent h for this wave's 4 column tiles x 2 M-subtiles

  for (int dir = 0; dir < 2; ++dir) {
    const unsigned short* Wih = dir ? Wihb : Wihf;
    const unsigned short* Whh = dir ? Whhb : Whhf;
    const float* bih = dir ? bihb : bihf;
    const float* bhh = dir ? bhhb : bhhf;

    #pragma unroll
    for (int ct = 0; ct < 4; ++ct)
      #pragma unroll
      for (int s = 0; s < 2; ++s)
        #pragma unroll
        for (int r = 0; r < 8; ++r) hreg[ct][s][r] = 0.0f;

    for (int i = tid * 8; i < 16384; i += 256 * 8) {
      v4u z = {0u, 0u, 0u, 0u};
      *(v4u*)(hbf0 + i) = z;
    }
    __syncthreads();

    for (int t = 0; t < MAXL; ++t) {
      { // cooperative gather of x_t (32 rows x 512 f32 -> bf16 LDS)
        int row = tid >> 3;
        int c0  = (tid & 7) * 64;
        int L   = dir ? lenB_s[row] : lenF_s[row];
        int idx = dir ? (6 + L - t) : (8 - L + t);   // in [0,14] by construction
        const float* src = pw + ((size_t)(gb + row) * T_WIN + idx) * D_IN + c0;
        unsigned short* dst = xs + row * D_IN + c0;
        #pragma unroll
        for (int j = 0; j < 64; j += 4) {
          float4 v = *(const float4*)(src + j);
          v2u pk;
          pk.x = (unsigned)f2bf(v.x) | ((unsigned)f2bf(v.y) << 16);
          pk.y = (unsigned)f2bf(v.z) | ((unsigned)f2bf(v.w) << 16);
          *(v2u*)(dst + j) = pk;
        }
      }
      __syncthreads();

      const unsigned short* hcur = (t & 1) ? hbf1 : hbf0;
      unsigned short*       hnxt = (t & 1) ? hbf0 : hbf1;

      #pragma unroll
      for (int ct = 0; ct < 4; ++ct) {
        const int colB = wave * 64 + ct * 16;
        const int n = colB + laneN;
        v8f accR[2], accZ[2], accXN[2], accHN[2];
        {
          accR[0]  = splat8(bih[n]        + bhh[n]);        accR[1]  = accR[0];
          accZ[0]  = splat8(bih[512 + n]  + bhh[512 + n]);  accZ[1]  = accZ[0];
          accXN[0] = splat8(bih[1024 + n]);                 accXN[1] = accXN[0];
          accHN[0] = splat8(bhh[1024 + n]);                 accHN[1] = accHN[0];
        }
        #pragma unroll 2
        for (int k0 = 0; k0 < 512; k0 += 32) {
          v16bf ax0 = loadA(xs,   laneN,      512, k0 + hi8);
          v16bf ax1 = loadA(xs,   16 + laneN, 512, k0 + hi8);
          v16bf ah0 = loadA(hcur, laneN,      512, k0 + hi8);
          v16bf ah1 = loadA(hcur, 16 + laneN, 512, k0 + hi8);
          v16bf b;
          b = loadB(Wih, n,        512, k0 + hi8);
          accR[0]  = wmma_bf(ax0, b, accR[0]);  accR[1]  = wmma_bf(ax1, b, accR[1]);
          b = loadB(Wih, 512 + n,  512, k0 + hi8);
          accZ[0]  = wmma_bf(ax0, b, accZ[0]);  accZ[1]  = wmma_bf(ax1, b, accZ[1]);
          b = loadB(Wih, 1024 + n, 512, k0 + hi8);
          accXN[0] = wmma_bf(ax0, b, accXN[0]); accXN[1] = wmma_bf(ax1, b, accXN[1]);
          b = loadB(Whh, n,        512, k0 + hi8);
          accR[0]  = wmma_bf(ah0, b, accR[0]);  accR[1]  = wmma_bf(ah1, b, accR[1]);
          b = loadB(Whh, 512 + n,  512, k0 + hi8);
          accZ[0]  = wmma_bf(ah0, b, accZ[0]);  accZ[1]  = wmma_bf(ah1, b, accZ[1]);
          b = loadB(Whh, 1024 + n, 512, k0 + hi8);
          accHN[0] = wmma_bf(ah0, b, accHN[0]); accHN[1] = wmma_bf(ah1, b, accHN[1]);
        }
        // gate combine: C-layout element (m,n): m = s*16 + hi*8 + r, n fixed per lane
        #pragma unroll
        for (int s = 0; s < 2; ++s) {
          #pragma unroll
          for (int r = 0; r < 8; ++r) {
            int row = s * 16 + hi8 + r;
            float rg = sigmf(accR[s][r]);
            float zg = sigmf(accZ[s][r]);
            float ng = tanhf(accXN[s][r] + rg * accHN[s][r]);
            float ho = hreg[ct][s][r];
            float hn = (1.0f - zg) * ng + zg * ho;
            int L = dir ? lenB_s[row] : lenF_s[row];
            float hv = (t < L) ? hn : ho;
            hreg[ct][s][r] = hv;
            hnxt[row * 512 + n] = f2bf(hv);
          }
        }
      }
      __syncthreads();
    }

    if (dir == 0) { // stash h_f as bf16 (read only in MLP phase)
      #pragma unroll
      for (int ct = 0; ct < 4; ++ct) {
        const int colB = wave * 64 + ct * 16;
        #pragma unroll
        for (int s = 0; s < 2; ++s)
          #pragma unroll
          for (int r = 0; r < 8; ++r) {
            int row = s * 16 + hi8 + r;
            hfb[row * 512 + colB + laneN] = f2bf(hreg[ct][s][r]);
          }
      }
    }
  }

  // ---------------- MLP ----------------
  __syncthreads();
  // comb[:, 0:512) = h_f (vector copy from hfb), comb[:, 512:1024) = h_b (regs)
  for (int j = tid; j < 2048; j += 256) {
    int row = j >> 6;
    ((v4u*)comb)[(size_t)row * 128 + (j & 63)] = ((const v4u*)hfb)[j];
  }
  #pragma unroll
  for (int ct = 0; ct < 4; ++ct) {
    const int colB = wave * 64 + ct * 16;
    #pragma unroll
    for (int s = 0; s < 2; ++s)
      #pragma unroll
      for (int r = 0; r < 8; ++r) {
        int row = s * 16 + hi8 + r;
        comb[row * 1024 + 512 + colB + laneN] = f2bf(hreg[ct][s][r]);
      }
  }
  __syncthreads();

  // hidden = relu(comb @ W1^T + b1)
  #pragma unroll
  for (int ct = 0; ct < 4; ++ct) {
    const int colB = wave * 64 + ct * 16;
    const int n = colB + laneN;
    v8f acc[2]; acc[0] = splat8(b1[n]); acc[1] = acc[0];
    #pragma unroll 2
    for (int k0 = 0; k0 < 1024; k0 += 32) {
      v16bf a0 = loadA(comb, laneN,      1024, k0 + hi8);
      v16bf a1 = loadA(comb, 16 + laneN, 1024, k0 + hi8);
      v16bf b  = loadB(W1b, n, 1024, k0 + hi8);
      acc[0] = wmma_bf(a0, b, acc[0]); acc[1] = wmma_bf(a1, b, acc[1]);
    }
    #pragma unroll
    for (int s = 0; s < 2; ++s)
      #pragma unroll
      for (int r = 0; r < 8; ++r) {
        int row = s * 16 + hi8 + r;
        float v = acc[s][r]; v = v > 0.0f ? v : 0.0f;
        hid[row * 512 + n] = f2bf(v);
      }
  }
  __syncthreads();

  // out = hidden @ W2^T + b2  (f32 stores to global)
  #pragma unroll
  for (int ct = 0; ct < 4; ++ct) {
    const int colB = wave * 64 + ct * 16;
    const int n = colB + laneN;
    v8f acc[2]; acc[0] = splat8(b2[n]); acc[1] = acc[0];
    #pragma unroll 2
    for (int k0 = 0; k0 < 512; k0 += 32) {
      v16bf a0 = loadA(hid, laneN,      512, k0 + hi8);
      v16bf a1 = loadA(hid, 16 + laneN, 512, k0 + hi8);
      v16bf b  = loadB(W2b, n, 512, k0 + hi8);
      acc[0] = wmma_bf(a0, b, acc[0]); acc[1] = wmma_bf(a1, b, acc[1]);
    }
    #pragma unroll
    for (int s = 0; s < 2; ++s)
      #pragma unroll
      for (int r = 0; r < 8; ++r) {
        int row = s * 16 + hi8 + r;
        out[(size_t)(gb + row) * 512 + n] = acc[s][r];
      }
  }
}

extern "C" void kernel_launch(void* const* d_in, const int* in_sizes, int n_in,
                              void* d_out, int out_size, void* d_ws, size_t ws_size,
                              hipStream_t stream) {
  (void)in_sizes; (void)n_in; (void)out_size; (void)ws_size;
  const float* pw   = (const float*)d_in[0];
  const int*   wlen = (const int*)d_in[1];
  const float* Wihf = (const float*)d_in[2];
  const float* Whhf = (const float*)d_in[3];
  const float* bihf = (const float*)d_in[4];
  const float* bhhf = (const float*)d_in[5];
  const float* Wihb = (const float*)d_in[6];
  const float* Whhb = (const float*)d_in[7];
  const float* bihb = (const float*)d_in[8];
  const float* bhhb = (const float*)d_in[9];
  const float* W1   = (const float*)d_in[10];
  const float* b1   = (const float*)d_in[11];
  const float* W2   = (const float*)d_in[12];
  const float* b2   = (const float*)d_in[13];
  float* out = (float*)d_out;

  // bf16 weight staging in workspace (~7.9 MB)
  unsigned short* ws = (unsigned short*)d_ws;
  unsigned short* Wihf_b = ws;
  unsigned short* Whhf_b = ws + 786432;
  unsigned short* Wihb_b = ws + 1572864;
  unsigned short* Whhb_b = ws + 2359296;
  unsigned short* W1_b   = ws + 3145728;
  unsigned short* W2_b   = ws + 3670016;

  auto cvt = [&](const float* s, unsigned short* d, int n) {
    cvt_bf16_kernel<<<(n + 255) / 256, 256, 0, stream>>>(s, d, n);
  };
  cvt(Wihf, Wihf_b, 786432);
  cvt(Whhf, Whhf_b, 786432);
  cvt(Wihb, Wihb_b, 786432);
  cvt(Whhb, Whhb_b, 786432);
  cvt(W1,   W1_b,   524288);
  cvt(W2,   W2_b,   262144);

  bigru_fused<<<B_TOT / MTILE, 256, 0, stream>>>(
      pw, wlen, Wihf_b, Whhf_b, Wihb_b, Whhb_b, W1_b, W2_b,
      bihf, bhhf, bihb, bhhb, b1, b2, out);
}
